// LieTransformerNet_37881611551187
// MI455X (gfx1250) — matrix operand order repaired and checked by reading, not compiled, not run on hardware
//
#include <hip/hip_runtime.h>
#include <hip/hip_bf16.h>
#include <stdint.h>

// ---------------- problem constants (from reference) ----------------
#define BB     2
#define NN     2304          // T*H*W
#define CC     96
#define NHEADS 4
#define HD     24
#define HIDN   255           // int(96*2.66)
#define HID2   510
#define HH     48
#define WW     48
#define NPIX   (HH*WW)       // 2304
#define MTOT   (BB*NN)       // 4608
#define ATT_SCALE 0.20412414523193154f   // HD^-0.5

typedef __attribute__((ext_vector_type(16))) _Float16 v16h;
typedef __attribute__((ext_vector_type(8)))  _Float16 v8h;
typedef __attribute__((ext_vector_type(8)))  float    v8f;

// ---------------- WMMA fragment loaders (per CDNA5 ISA layouts) ---
// A-matrix 16x32 f16: lane l -> row m = l&15 ; lanes 0-15 hold K 0-7 & 16-23,
// lanes 16-31 hold K 8-15 & 24-31 (two halves per VGPR).
__device__ __forceinline__ v16h frag_a_global(const _Float16* __restrict__ p,
                                              int ld, int row0, int k0) {
  int l  = threadIdx.x & 31;
  int m  = l & 15;
  int kb = (l >> 4) * 8;
  const _Float16* r = p + (size_t)(row0 + m) * ld + k0;
  v8h lo = *(const v8h*)(r + kb);
  v8h hi = *(const v8h*)(r + kb + 16);
  v16h a;
#pragma unroll
  for (int i = 0; i < 8; ++i) { a[i] = lo[i]; a[8 + i] = hi[i]; }
  return a;
}

// B-matrix 32x16 f16 loaded from an n-major buffer (each output column n has
// its K values contiguous): lane l -> col n = l&15 ; lanes 0-15 hold K 0-7 & 16-23,
// lanes 16-31 hold K 8-15 & 24-31.
__device__ __forceinline__ v16h frag_b_nmajor(const _Float16* __restrict__ p,
                                              int ldk, int col0, int k0) {
  int l  = threadIdx.x & 31;
  int n  = l & 15;
  int kb = (l >> 4) * 8;
  const _Float16* r = p + (size_t)(col0 + n) * ldk + k0 + kb;
  v8h lo = *(const v8h*)(r);
  v8h hi = *(const v8h*)(r + 16);
  v16h b;
#pragma unroll
  for (int i = 0; i < 8; ++i) { b[i] = lo[i]; b[8 + i] = hi[i]; }
  return b;
}

// A-fragment from a 16x32 f16 tile in LDS (emits ds_load_b128)
__device__ __forceinline__ v16h frag_a_lds(const _Float16* p) {
  int l  = threadIdx.x & 31;
  int m  = l & 15;
  int kb = (l >> 4) * 8;
  const _Float16* r = p + m * 32;
  v8h lo = *(const v8h*)(r + kb);
  v8h hi = *(const v8h*)(r + kb + 16);
  v16h a;
#pragma unroll
  for (int i = 0; i < 8; ++i) { a[i] = lo[i]; a[8 + i] = hi[i]; }
  return a;
}

// ---------------- weight convert + pad (f32 -> f16, zero pad) ----------------
__global__ void k_cvt_pad(const float* __restrict__ src, _Float16* __restrict__ dst,
                          int rows, int cols, int prows, int pcols) {
  int i = blockIdx.x * blockDim.x + threadIdx.x;
  int tot = prows * pcols;
  if (i >= tot) return;
  int r = i / pcols, c = i % pcols;
  float v = (r < rows && c < cols) ? src[(size_t)r * cols + c] : 0.f;
  dst[i] = (_Float16)v;
}

// ---------------- LayerNorm (one wave per token, C = 96 = 3 lanes-strided) ---
__global__ void k_ln(const float* __restrict__ x, const float* __restrict__ w,
                     const float* __restrict__ b, _Float16* __restrict__ y, int ntok) {
  int wave = threadIdx.x >> 5, lane = threadIdx.x & 31;
  int tok = blockIdx.x * 8 + wave;
  if (tok >= ntok) return;
  const float* xr = x + (size_t)tok * CC;
  float v0 = xr[lane], v1 = xr[lane + 32], v2 = xr[lane + 64];
  float s = v0 + v1 + v2, sq = v0 * v0 + v1 * v1 + v2 * v2;
#pragma unroll
  for (int off = 1; off < 32; off <<= 1) {
    s  += __shfl_xor(s,  off, 32);
    sq += __shfl_xor(sq, off, 32);
  }
  float m  = s * (1.f / CC);
  float var = sq * (1.f / CC) - m * m;
  float rs = rsqrtf(var + 1e-5f);
  _Float16* yr = y + (size_t)tok * CC;
  yr[lane]      = (_Float16)((v0 - m) * rs * w[lane]      + b[lane]);
  yr[lane + 32] = (_Float16)((v1 - m) * rs * w[lane + 32] + b[lane + 32]);
  yr[lane + 64] = (_Float16)((v2 - m) * rs * w[lane + 64] + b[lane + 64]);
}

// ------------- residual (x + gamma*res) then LayerNorm, both outputs --------
__global__ void k_resid_ln(const float* __restrict__ x, const float* __restrict__ res,
                           const float* __restrict__ gamma,
                           const float* __restrict__ w, const float* __restrict__ b,
                           float* __restrict__ x1, _Float16* __restrict__ y, int ntok) {
  int wave = threadIdx.x >> 5, lane = threadIdx.x & 31;
  int tok = blockIdx.x * 8 + wave;
  if (tok >= ntok) return;
  size_t base = (size_t)tok * CC;
  float v0 = x[base + lane]      + gamma[lane]      * res[base + lane];
  float v1 = x[base + lane + 32] + gamma[lane + 32] * res[base + lane + 32];
  float v2 = x[base + lane + 64] + gamma[lane + 64] * res[base + lane + 64];
  x1[base + lane] = v0; x1[base + lane + 32] = v1; x1[base + lane + 64] = v2;
  float s = v0 + v1 + v2, sq = v0 * v0 + v1 * v1 + v2 * v2;
#pragma unroll
  for (int off = 1; off < 32; off <<= 1) {
    s  += __shfl_xor(s,  off, 32);
    sq += __shfl_xor(sq, off, 32);
  }
  float m  = s * (1.f / CC);
  float var = sq * (1.f / CC) - m * m;
  float rs = rsqrtf(var + 1e-5f);
  y[base + lane]      = (_Float16)((v0 - m) * rs * w[lane]      + b[lane]);
  y[base + lane + 32] = (_Float16)((v1 - m) * rs * w[lane + 32] + b[lane + 32]);
  y[base + lane + 64] = (_Float16)((v2 - m) * rs * w[lane + 64] + b[lane + 64]);
}

// ---------------- generic WMMA GEMM: C[M,N] = A[M,Kp](f16) * W[N,Kp](f16)^T + bias
// A row-major (lda=Kp), W n-major padded (rows = ceil16(N), cols = Kp mult-32).
// One 16x16 output tile per wave, 8 waves per block.
__global__ void k_gemm(const _Float16* __restrict__ A, int lda,
                       const _Float16* __restrict__ Bw, int ldb,
                       const float* __restrict__ bias,
                       float* __restrict__ Cout, int ldc,
                       int M, int Nout, int Kp) {
  int wave = threadIdx.x >> 5;
  int tile = blockIdx.x * (blockDim.x >> 5) + wave;
  int tn_cnt = (Nout + 15) >> 4;
  int tm_cnt = M >> 4;
  if (tile >= tm_cnt * tn_cnt) return;
  int tm = tile / tn_cnt, tn = tile % tn_cnt;

  v8f acc = {};
  for (int k0 = 0; k0 < Kp; k0 += 32) {
    v16h a = frag_a_global(A, lda, tm * 16, k0);
    v16h b = frag_b_nmajor(Bw, ldb, tn * 16, k0);
    acc = __builtin_amdgcn_wmma_f32_16x16x32_f16(false, a, false, b,
                                                 (short)0, acc, false, false);
  }
  int l = threadIdx.x & 31;
  int n = l & 15;
  int mh = (l >> 4) * 8;
#pragma unroll
  for (int r = 0; r < 8; ++r) {
    int col = tn * 16 + n;
    if (col < Nout)
      Cout[(size_t)(tm * 16 + mh + r) * ldc + col] = acc[r] + (bias ? bias[col] : 0.f);
  }
}

// -------- scatter qkv f32 -> padded Q/K (B,H,N,32) f16 and V^T (B,H,32,N) ----
__global__ void k_qkv_scatter(const float* __restrict__ qkv,
                              _Float16* __restrict__ Qb, _Float16* __restrict__ Kb,
                              _Float16* __restrict__ Vt) {
  int i = blockIdx.x * blockDim.x + threadIdx.x;      // B*H*N*32
  if (i >= BB * NHEADS * NN * 32) return;
  int d  = i & 31;
  int n  = (i >> 5) % NN;
  int bh = i / (32 * NN);
  int b  = bh / NHEADS, h = bh % NHEADS;
  size_t row = (size_t)(b * NN + n) * (3 * CC);
  float q = 0.f, k = 0.f, v = 0.f;
  if (d < HD) {
    int c = h * HD + d;
    q = qkv[row + c];
    k = qkv[row + CC + c];
    v = qkv[row + 2 * CC + c];
  }
  Qb[i] = (_Float16)q;
  Kb[i] = (_Float16)k;
  Vt[(size_t)(bh * 32 + d) * NN + n] = (_Float16)v;
}

// ---------------- flash attention with fused bias-MLP ------------------------
// 1 wave per (b, h, 16-query tile). S via 2x WMMA per 32-key chunk, bias MLP
// fused into score tile in C/D layout, online softmax, P bounced through LDS
// to A layout, PV via 2x WMMA into persistent f32 accumulators.
__global__ void __launch_bounds__(32)
k_flash(const _Float16* __restrict__ Qb, const _Float16* __restrict__ Kb,
        const _Float16* __restrict__ Vt, const float* __restrict__ bias,
        const float* __restrict__ m1w, const float* __restrict__ m1b,
        const float* __restrict__ m2w, const float* __restrict__ m2b,
        _Float16* __restrict__ Oout) {
  __shared__ _Float16 pbuf[16 * 32];

  int q0 = blockIdx.x * 16;
  int b  = blockIdx.y / NHEADS;
  int h  = blockIdx.y % NHEADS;
  int l  = threadIdx.x & 31;
  int n  = l & 15;
  int mh = (l >> 4) * 8;

  const _Float16* Qbase  = Qb + (size_t)(b * NHEADS + h) * NN * 32;
  const _Float16* Kbase  = Kb + (size_t)(b * NHEADS + h) * NN * 32;
  const _Float16* Vtbase = Vt + (size_t)(b * NHEADS + h) * 32 * NN;
  const float*    brow   = bias + (size_t)b * NN * NN;

  float w1[16], wb1[16], w2[16];
#pragma unroll
  for (int j = 0; j < 16; ++j) {
    w1[j]  = m1w[j];
    wb1[j] = m1b[j];
    w2[j]  = m2w[h * 16 + j];
  }
  float b2 = m2b[h];

  v16h aq = frag_a_global(Qbase, 32, q0, 0);   // 16x32 (HD padded to 32)

  float mrow[8], lrow[8];
  v8f o0 = {}, o1 = {};
#pragma unroll
  for (int r = 0; r < 8; ++r) { mrow[r] = -1e30f; lrow[r] = 0.f; }

  for (int kk = 0; kk < NN; kk += 32) {
    v16h bk0 = frag_b_nmajor(Kbase, 32, kk,      0);
    v16h bk1 = frag_b_nmajor(Kbase, 32, kk + 16, 0);
    v8f z = {};
    v8f s0 = __builtin_amdgcn_wmma_f32_16x16x32_f16(false, aq, false, bk0,
                                                    (short)0, z, false, false);
    v8f s1 = __builtin_amdgcn_wmma_f32_16x16x32_f16(false, aq, false, bk1,
                                                    (short)0, z, false, false);
    if (kk + 32 < NN) {                              // global_prefetch_b8
      __builtin_prefetch(Kbase + (size_t)(kk + 32) * 32, 0, 1);
      __builtin_prefetch(Vtbase + kk + 32, 0, 1);
    }

    // fused learned-bias MLP: lb = relu(bias*m1w + m1b) @ m2w[h] + m2b[h]
#pragma unroll
    for (int r = 0; r < 8; ++r) {
      const float* bp = brow + (size_t)(q0 + mh + r) * NN + kk + n;
      float bv0 = bp[0], bv1 = bp[16];
      float a0 = b2, a1 = b2;
#pragma unroll
      for (int j = 0; j < 16; ++j) {
        float t0 = fmaxf(fmaf(bv0, w1[j], wb1[j]), 0.f);
        float t1 = fmaxf(fmaf(bv1, w1[j], wb1[j]), 0.f);
        a0 = fmaf(t0, w2[j], a0);
        a1 = fmaf(t1, w2[j], a1);
      }
      s0[r] = s0[r] * ATT_SCALE + a0;
      s1[r] = s1[r] * ATT_SCALE + a1;
    }

    // online softmax (row reductions within 16-lane halves)
#pragma unroll
    for (int r = 0; r < 8; ++r) {
      float mx = fmaxf(s0[r], s1[r]);
#pragma unroll
      for (int off = 1; off < 16; off <<= 1) mx = fmaxf(mx, __shfl_xor(mx, off, 32));
      float mnew  = fmaxf(mrow[r], mx);
      float alpha = __expf(mrow[r] - mnew);
      float p0 = __expf(s0[r] - mnew);
      float p1 = __expf(s1[r] - mnew);
      float sum = p0 + p1;
#pragma unroll
      for (int off = 1; off < 16; off <<= 1) sum += __shfl_xor(sum, off, 32);
      lrow[r] = lrow[r] * alpha + sum;
      mrow[r] = mnew;
      o0[r] *= alpha;
      o1[r] *= alpha;
      pbuf[(mh + r) * 32 + n]      = (_Float16)p0;   // ds_store_b16
      pbuf[(mh + r) * 32 + 16 + n] = (_Float16)p1;
    }
    __syncthreads();
    v16h ap = frag_a_lds(pbuf);                      // ds_load_b128
    __syncthreads();

    v16h bv0 = frag_b_nmajor(Vtbase, NN, 0,  kk);    // dims 0-15
    v16h bv1 = frag_b_nmajor(Vtbase, NN, 16, kk);    // dims 16-23 (+pad)
    o0 = __builtin_amdgcn_wmma_f32_16x16x32_f16(false, ap, false, bv0,
                                                (short)0, o0, false, false);
    o1 = __builtin_amdgcn_wmma_f32_16x16x32_f16(false, ap, false, bv1,
                                                (short)0, o1, false, false);
  }

#pragma unroll
  for (int r = 0; r < 8; ++r) {
    float inv = lrow[r] > 0.f ? 1.f / lrow[r] : 0.f;
    size_t rb = (size_t)(b * NN + q0 + mh + r) * CC + h * HD;
    Oout[rb + n] = (_Float16)(o0[r] * inv);
    if (n < 8) Oout[rb + 16 + n] = (_Float16)(o1[r] * inv);
  }
}

// --------- depthwise 3x3 conv + GELU gate: z = gelu(z1)*z2, f16 out ----------
__global__ void k_dwgate(const float* __restrict__ zin,   // [4608][512] (510 valid)
                         const float* __restrict__ dw_w,  // [510][9]
                         const float* __restrict__ dw_b,  // [510]
                         _Float16* __restrict__ gate) {   // [4608][256] (255 valid)
  int i = blockIdx.x * blockDim.x + threadIdx.x;
  if (i >= MTOT * 256) return;
  int ch  = i & 255;
  int pix = i >> 8;
  if (ch == 255) { gate[(size_t)pix * 256 + 255] = (_Float16)0.f; return; }
  int img = pix / NPIX;
  int p   = pix % NPIX;
  int y = p / WW, x = p % WW;
  int c1 = ch, c2 = ch + HIDN;
  float a1 = dw_b[c1], a2 = dw_b[c2];
#pragma unroll
  for (int dy = -1; dy <= 1; ++dy)
#pragma unroll
    for (int dx = -1; dx <= 1; ++dx) {
      int yy = y + dy, xx = x + dx;
      if (yy < 0 || yy >= HH || xx < 0 || xx >= WW) continue;
      size_t off = (size_t)(img * NPIX + yy * WW + xx) * 512;
      int t = (dy + 1) * 3 + (dx + 1);
      a1 = fmaf(zin[off + c1], dw_w[c1 * 9 + t], a1);
      a2 = fmaf(zin[off + c2], dw_w[c2 * 9 + t], a2);
    }
  float g = 0.5f * a1 * (1.f + erff(a1 * 0.70710678f));  // exact gelu
  gate[(size_t)pix * 256 + ch] = (_Float16)(g * a2);
}

// ---------------- final residual ---------------------------------------------
__global__ void k_final(const float* __restrict__ x1, const float* __restrict__ g2,
                        const float* __restrict__ pout, float* __restrict__ out) {
  int i = blockIdx.x * blockDim.x + threadIdx.x;
  if (i >= MTOT * CC) return;
  out[i] = x1[i] + g2[i % CC] * pout[i];
}

// =============================================================================
extern "C" void kernel_launch(void* const* d_in, const int* in_sizes, int n_in,
                              void* d_out, int out_size, void* d_ws, size_t ws_size,
                              hipStream_t stream) {
  const float* x       = (const float*)d_in[0];
  const float* bias    = (const float*)d_in[1];
  const float* norm1_w = (const float*)d_in[2];
  const float* norm1_b = (const float*)d_in[3];
  const float* norm2_w = (const float*)d_in[4];
  const float* norm2_b = (const float*)d_in[5];
  const float* qkv_w   = (const float*)d_in[6];
  const float* qkv_b   = (const float*)d_in[7];
  const float* proj_w  = (const float*)d_in[8];
  const float* proj_b  = (const float*)d_in[9];
  const float* m1w     = (const float*)d_in[10];
  const float* m1b     = (const float*)d_in[11];
  const float* m2w     = (const float*)d_in[12];
  const float* m2b     = (const float*)d_in[13];
  const float* pin_w   = (const float*)d_in[14];
  const float* pin_b   = (const float*)d_in[15];
  const float* dw_w    = (const float*)d_in[16];
  const float* dw_b    = (const float*)d_in[17];
  const float* pout_w  = (const float*)d_in[18];
  const float* pout_b  = (const float*)d_in[19];
  const float* gamma1  = (const float*)d_in[20];
  const float* gamma2  = (const float*)d_in[21];
  float* out = (float*)d_out;

  uint8_t* ws = (uint8_t*)d_ws;
  size_t off = 0;
  auto alloc = [&](size_t bytes) -> void* {
    void* p = ws + off;
    off = (off + bytes + 255) & ~(size_t)255;
    return p;
  };

  _Float16* xn1   = (_Float16*)alloc((size_t)MTOT * CC * 2);
  float*    qkvf  = (float*)   alloc((size_t)MTOT * 3 * CC * 4);
  _Float16* Qb    = (_Float16*)alloc((size_t)BB * NHEADS * NN * 32 * 2);
  _Float16* Kbuf  = (_Float16*)alloc((size_t)BB * NHEADS * NN * 32 * 2);
  _Float16* Vt    = (_Float16*)alloc((size_t)BB * NHEADS * 32 * NN * 2);
  _Float16* attn  = (_Float16*)alloc((size_t)MTOT * CC * 2);
  float*    projf = (float*)   alloc((size_t)MTOT * CC * 4);
  float*    x1    = (float*)   alloc((size_t)MTOT * CC * 4);
  _Float16* xn2   = (_Float16*)alloc((size_t)MTOT * CC * 2);
  float*    pinf  = (float*)   alloc((size_t)MTOT * 512 * 4);
  _Float16* gatef = (_Float16*)alloc((size_t)MTOT * 256 * 2);
  float*    poutf = (float*)   alloc((size_t)MTOT * CC * 4);
  _Float16* Wq    = (_Float16*)alloc((size_t)288 * 96 * 2);
  _Float16* Wproj = (_Float16*)alloc((size_t)96 * 96 * 2);
  _Float16* Wpin  = (_Float16*)alloc((size_t)512 * 96 * 2);
  _Float16* Wpout = (_Float16*)alloc((size_t)96 * 256 * 2);
  (void)ws_size; (void)in_sizes; (void)n_in; (void)out_size;

  // 1) weight conversion (f16, zero-padded to WMMA tile multiples)
  k_cvt_pad<<<(288 * 96 + 255) / 256, 256, 0, stream>>>(qkv_w,  Wq,    288, 96,  288, 96);
  k_cvt_pad<<<(96 * 96 + 255) / 256,  256, 0, stream>>>(proj_w, Wproj, 96,  96,  96,  96);
  k_cvt_pad<<<(512 * 96 + 255) / 256, 256, 0, stream>>>(pin_w,  Wpin,  510, 96,  512, 96);
  k_cvt_pad<<<(96 * 256 + 255) / 256, 256, 0, stream>>>(pout_w, Wpout, 96,  255, 96,  256);

  // 2) LN1 -> qkv GEMM -> scatter to padded Q/K/V^T
  k_ln<<<MTOT / 8, 256, 0, stream>>>(x, norm1_w, norm1_b, xn1, MTOT);
  {
    int tiles = (MTOT / 16) * (288 / 16);
    k_gemm<<<(tiles + 7) / 8, 256, 0, stream>>>(xn1, 96, Wq, 96, qkv_b,
                                                qkvf, 288, MTOT, 288, 96);
  }
  k_qkv_scatter<<<(BB * NHEADS * NN * 32) / 256, 256, 0, stream>>>(qkvf, Qb, Kbuf, Vt);

  // 3) flash attention, fused bias MLP
  {
    dim3 grid(NN / 16, BB * NHEADS);
    k_flash<<<grid, 32, 0, stream>>>(Qb, Kbuf, Vt, bias, m1w, m1b, m2w, m2b, attn);
  }

  // 4) proj GEMM -> residual + LN2
  {
    int tiles = (MTOT / 16) * (96 / 16);
    k_gemm<<<(tiles + 7) / 8, 256, 0, stream>>>(attn, 96, Wproj, 96, proj_b,
                                                projf, 96, MTOT, 96, 96);
  }
  k_resid_ln<<<MTOT / 8, 256, 0, stream>>>(x, projf, gamma1, norm2_w, norm2_b,
                                           x1, xn2, MTOT);

  // 5) GDFN: pin GEMM -> depthwise+gate -> pout GEMM -> residual
  {
    int tiles = (MTOT / 16) * ((510 + 15) / 16);
    k_gemm<<<(tiles + 7) / 8, 256, 0, stream>>>(xn2, 96, Wpin, 96, pin_b,
                                                pinf, 512, MTOT, 510, 96);
  }
  k_dwgate<<<(MTOT * 256) / 256, 256, 0, stream>>>(pinf, dw_w, dw_b, gatef);
  {
    int tiles = (MTOT / 16) * (96 / 16);
    k_gemm<<<(tiles + 7) / 8, 256, 0, stream>>>(gatef, 256, Wpout, 256, pout_b,
                                                poutf, 96, MTOT, 96, 256);
  }
  k_final<<<(MTOT * CC + 255) / 256, 256, 0, stream>>>(x1, gamma2, poutf, out);
}
